// LinearAttention_35158602285775
// MI455X (gfx1250) — matrix operand order repaired and checked by reading, not compiled, not run on hardware
//
#include <hip/hip_runtime.h>

// ---------------------------------------------------------------------------
// LinearAttention (B=16, N=4096, C=256, 8 heads x 32) for MI455X / gfx1250.
//   K0: pw = sigmoid(pos_weights gather)            (4 MB, L2-resident, once)
//   K1: partial kv = k_rope^T v over N/4 segments   (v_wmma_f32_16x16x32_f16)
//       + partial k_mean (ds_add_f32 LDS reduce)    512 blocks for occupancy
//   K1b: combine 4 segments, scale by 1/N
//   K2: attn = (q_rope @ kv) * z fused with depthwise-3x3 LePE + bias,
//       single float4 output write.
// Memory-bound (~270 MB min traffic @ 23.3 TB/s ~= 12 us); WMMA keeps the
// einsums off the VALU; sigmoid hoisting halves transcendental work.
// ---------------------------------------------------------------------------

#define HH   64
#define WWI  64
#define NN   4096
#define CC   256
#define NH   8
#define HD   32
#define PWW  65   // 2*DW+1 (DH=DW=32); DX_IDX = i+1, DY_IDX = j+1
#define BB   16
#define SEG  4    // N-segments for kernel-1 occupancy (512 blocks)

typedef __attribute__((ext_vector_type(16))) _Float16 v16h_t;
typedef __attribute__((ext_vector_type(8)))  float    v8f_t;

union HVec { v16h_t v; _Float16 h[16]; };
union FVec { v8f_t  v; float    f[8];  };

__device__ __forceinline__ float elu1(float x) {
  // elu(x)+1 : x>0 -> x+1 ; x<=0 -> exp(x)
  return x > 0.f ? x + 1.f : __expf(x);
}
__device__ __forceinline__ float sigm(float x) {
  return 1.f / (1.f + __expf(-x));
}

// ---------------------------------------------------------------------------
// K0: positional gate precompute, pw[n, c] = sigmoid(posw[i+1, j+1, c]).
// ---------------------------------------------------------------------------
__global__ __launch_bounds__(256)
void pw_precompute_kernel(const float* __restrict__ posw, float* __restrict__ pw) {
  const int n = blockIdx.x;
  const int c = threadIdx.x;
  const int i = n >> 6, j = n & 63;
  pw[(size_t)n * CC + c] = sigm(posw[((i + 1) * PWW + (j + 1)) * CC + c]);
}

// ---------------------------------------------------------------------------
// K1: per (b,h,seg) block, 256 threads = 8 waves.
//   wave w: eblock = w&1, dblock = (w>>1)&1, nhalf = w>>2
//   Each wave accumulates one 16x16 tile of kv over N/(SEG*2) rows with WMMA.
//   A (16x32 f16): rows = d (lane%16 + 16*dblock), K = n-chunk
//     element idx -> K = idx + (idx&8) + (lane>=16)*8      (ISA A-layout)
//   B (32x16 f16): cols = e (lane%16 + 16*eblock), K rows
//     element idx -> K = idx + (idx&8) + (lane>=16)*8      (ISA B-layout)
//   Partial k_mean (raw k) accumulated by eblock==0 waves via ds_add_f32.
// ---------------------------------------------------------------------------
__global__ __launch_bounds__(256)
void kv_partial_kernel(const float* __restrict__ x2, const float* __restrict__ x3,
                       const float* __restrict__ pw,
                       float* __restrict__ kv_part, float* __restrict__ km_part) {
  const int blk  = blockIdx.x;           // (b*8+h)*SEG + seg
  const int bh   = blk >> 2;
  const int seg  = blk & (SEG - 1);
  const int b    = bh >> 3;
  const int h    = bh & 7;
  const int t    = threadIdx.x;
  const int lane = t & 31;
  const int wid  = t >> 5;
  const int col  = lane & 15;
  const int hi   = lane >> 4;            // 0: lanes 0-15, 1: lanes 16-31
  const int dblock = (wid >> 1) & 1;
  const int eblock = wid & 1;
  const int nhalf  = wid >> 2;

  const int cA = h * HD + dblock * 16 + col;   // channel of kr column (A row d)
  const int cB = h * HD + eblock * 16 + col;   // channel of v  column (B col e)
  const size_t xb = (size_t)b * NN * CC;

  FVec acc;
  #pragma unroll
  for (int r = 0; r < 8; ++r) acc.f[r] = 0.f;
  float ksum = 0.f;

  const int nbeg = seg * (NN / SEG) + nhalf * (NN / SEG / 2);
  for (int n0 = nbeg; n0 < nbeg + NN / SEG / 2; n0 += 32) {
    HVec A, B;
    #pragma unroll
    for (int idx = 0; idx < 16; ++idx) {
      // ---- A element: kr[n0+kA, cA] ----
      const int kA = idx + (idx & 8) + (hi << 3);
      const int nA = n0 + kA;
      const float kk = elu1(x2[xb + (size_t)nA * CC + cA]);
      ksum += kk;                                   // used only by eblock==0
      A.h[idx] = (_Float16)(kk * pw[(size_t)nA * CC + cA]);
      // ---- B element: v[n0+kB, cB] ----
      const int kB = idx + (idx & 8) + (hi << 3);
      B.h[idx] = (_Float16)x3[xb + (size_t)(n0 + kB) * CC + cB];
    }
    acc.v = __builtin_amdgcn_wmma_f32_16x16x32_f16(false, A.v, false, B.v,
                                                   (short)0, acc.v, false, false);
  }

  __shared__ float kvs[8][32][8];
  __shared__ float ksum_s[HD];
  if (t < HD) ksum_s[t] = 0.f;
  __syncthreads();
  #pragma unroll
  for (int r = 0; r < 8; ++r) kvs[wid][lane][r] = acc.f[r];
  if (eblock == 0) atomicAdd(&ksum_s[dblock * 16 + col], ksum);
  __syncthreads();

  if (wid < 4) {   // combine the two n-halves, write unscaled partial kv
    const int dbase = ((wid >> 1) & 1) * 16;
    const int ebase = (wid & 1) * 16;
    #pragma unroll
    for (int r = 0; r < 8; ++r) {
      const float s = kvs[wid][lane][r] + kvs[wid + 4][lane][r];
      const int d = dbase + r + (hi << 3);          // D row: M = r + 8*hi
      const int e = ebase + col;                    // D col: N = lane%16
      kv_part[((size_t)blk * HD + d) * HD + e] = s;
    }
  }
  if (t < HD) km_part[(size_t)blk * HD + t] = ksum_s[t];
}

// ---------------------------------------------------------------------------
// K1b: combine SEG partial results, scale by 1/N.
//   131072 kv elements over 512 blocks x 256 threads; first 4096 threads also
//   combine k_mean.
// ---------------------------------------------------------------------------
__global__ __launch_bounds__(256)
void kv_combine_kernel(const float* __restrict__ kv_part,
                       const float* __restrict__ km_part,
                       float* __restrict__ kv_ws, float* __restrict__ km_ws) {
  const int g = blockIdx.x * 256 + threadIdx.x;      // 0 .. 131071
  const int bh  = g >> 10;                           // /1024
  const int rem = g & 1023;
  float s = 0.f;
  #pragma unroll
  for (int seg = 0; seg < SEG; ++seg)
    s += kv_part[((size_t)bh * SEG + seg) * (HD * HD) + rem];
  kv_ws[g] = s * (1.f / (float)NN);
  if (g < BB * NH * HD) {                            // 4096 k_mean values
    const int bh2 = g >> 5, d = g & 31;
    float m = 0.f;
    #pragma unroll
    for (int seg = 0; seg < SEG; ++seg)
      m += km_part[((size_t)bh2 * SEG + seg) * HD + d];
    km_ws[g] = m * (1.f / (float)NN);
  }
}

// ---------------------------------------------------------------------------
// K2: per block: one batch b, 16 pixels n0..n0+15 (same image row).
//   Phase A (8 waves = 8 heads): A = q_rope rows (16x32), B = kv (32x16) x2
//   eblocks -> 2 WMMAs; z = 1/(q . k_mean + 1e-6) via lane-split dot +
//   __shfl_xor(.,16); scaled result staged in LDS.
//   Phase B: remap thread->(pixel, 16 channels); depthwise 3x3 LePE from
//   LDS-staged weights, fuse with attn, single float4-vectorized store.
// ---------------------------------------------------------------------------
__global__ __launch_bounds__(256)
void attn_lepe_kernel(const float* __restrict__ x1, const float* __restrict__ x3,
                      const float* __restrict__ pw,
                      const float* __restrict__ lepe_w, const float* __restrict__ lepe_b,
                      const float* __restrict__ kv_ws, const float* __restrict__ km_ws,
                      float* __restrict__ out) {
  const int b    = blockIdx.x >> 8;      // 256 pixel-groups per batch
  const int g    = blockIdx.x & 255;
  const int n0   = g * 16;
  const int t    = threadIdx.x;
  const int lane = t & 31;
  const int wid  = t >> 5;               // == head index
  const int col  = lane & 15;
  const int hi   = lane >> 4;

  __shared__ float attn_s[16 * CC];      // 16 KB
  __shared__ float km_s[CC];             // k_mean for all heads of this batch
  __shared__ float w_s[CC * 9];          // depthwise weights, 9 KB
  __shared__ float bias_s[CC];

  km_s[t]   = km_ws[b * CC + t];         // (b*8+h)*32+d == b*256 + c
  bias_s[t] = lepe_b[t];
  for (int r = t; r < CC * 9; r += 256) w_s[r] = lepe_w[r];
  __syncthreads();

  // ---------------- attention (WMMA) ----------------
  {
    const int h = wid;
    HVec B0, B1, A;
    #pragma unroll
    for (int idx = 0; idx < 16; ++idx) {           // B: kv[d, e], K=d
      const int d = idx + (idx & 8) + (hi << 3);
      const size_t kb = ((size_t)(b * NH + h) * HD + d) * HD;
      B0.h[idx] = (_Float16)kv_ws[kb + col];
      B1.h[idx] = (_Float16)kv_ws[kb + 16 + col];
    }
    const int n  = n0 + col;                       // A row (M = lane%16)
    const size_t xrow = ((size_t)b * NN + n) * CC;
    const size_t prow = (size_t)n * CC;
    float zp = 0.f;
    #pragma unroll
    for (int idx = 0; idx < 16; ++idx) {           // A: q_rope[n, d], K=d
      const int d = idx + (idx & 8) + (hi << 3);
      const int c = wid * HD + d;
      const float q = elu1(x1[xrow + c]);
      zp += q * km_s[c];                           // ungated q . k_mean
      A.h[idx] = (_Float16)(q * pw[prow + c]);
    }
    const float zs = zp + __shfl_xor(zp, 16, 32);  // wave32: combine d-halves
    const float zi = 1.f / (zs + 1e-6f);

    FVec C0, C1;
    #pragma unroll
    for (int r = 0; r < 8; ++r) { C0.f[r] = 0.f; C1.f[r] = 0.f; }
    C0.v = __builtin_amdgcn_wmma_f32_16x16x32_f16(false, A.v, false, B0.v,
                                                  (short)0, C0.v, false, false);
    C1.v = __builtin_amdgcn_wmma_f32_16x16x32_f16(false, A.v, false, B1.v,
                                                  (short)0, C1.v, false, false);
    #pragma unroll
    for (int r = 0; r < 8; ++r) {                  // D row rr = r + 8*hi
      const int rr = r + (hi << 3);
      const float zr = __shfl(zi, rr, 32);         // z for output row rr
      attn_s[rr * CC + h * HD + col]      = C0.f[r] * zr;
      attn_s[rr * CC + h * HD + 16 + col] = C1.f[r] * zr;
    }
  }
  __syncthreads();

  // ---------------- LePE + fuse + store ----------------
  {
    const int px = t >> 4, cg = t & 15;
    const int c0 = cg * 16;
    const int n  = n0 + px;
    const int ii = n >> 6, jj = n & 63;
    float accv[16];
    #pragma unroll
    for (int cc = 0; cc < 16; ++cc) accv[cc] = bias_s[c0 + cc];
    #pragma unroll
    for (int di = -1; di <= 1; ++di) {
      const int ri = ii + di;
      if ((unsigned)ri >= (unsigned)HH) continue;
      #pragma unroll
      for (int dj = -1; dj <= 1; ++dj) {
        const int rj = jj + dj;
        if ((unsigned)rj >= (unsigned)WWI) continue;
        const int widx = (di + 1) * 3 + (dj + 1);
        const float4* vp =
            (const float4*)(x3 + (((size_t)b * NN + ri * WWI + rj) * CC + c0));
        #pragma unroll
        for (int q4 = 0; q4 < 4; ++q4) {
          const float4 vv = vp[q4];
          accv[q4*4+0] += vv.x * w_s[(c0 + q4*4+0) * 9 + widx];
          accv[q4*4+1] += vv.y * w_s[(c0 + q4*4+1) * 9 + widx];
          accv[q4*4+2] += vv.z * w_s[(c0 + q4*4+2) * 9 + widx];
          accv[q4*4+3] += vv.w * w_s[(c0 + q4*4+3) * 9 + widx];
        }
      }
    }
    float4* op = (float4*)(out + (((size_t)b * NN + n) * CC + c0));
    #pragma unroll
    for (int q4 = 0; q4 < 4; ++q4) {
      float4 o;
      o.x = attn_s[px * CC + c0 + q4*4+0] + accv[q4*4+0];
      o.y = attn_s[px * CC + c0 + q4*4+1] + accv[q4*4+1];
      o.z = attn_s[px * CC + c0 + q4*4+2] + accv[q4*4+2];
      o.w = attn_s[px * CC + c0 + q4*4+3] + accv[q4*4+3];
      op[q4] = o;
    }
  }
}

extern "C" void kernel_launch(void* const* d_in, const int* in_sizes, int n_in,
                              void* d_out, int out_size, void* d_ws, size_t ws_size,
                              hipStream_t stream) {
  const float* x1   = (const float*)d_in[0];
  const float* x2   = (const float*)d_in[1];
  const float* x3   = (const float*)d_in[2];
  const float* posw = (const float*)d_in[3];
  const float* lw   = (const float*)d_in[4];
  const float* lb   = (const float*)d_in[5];
  float* out = (float*)d_out;

  // Workspace layout (floats): total ~6.9 MB, fully rewritten each call.
  float* pw_ws   = (float*)d_ws;                               // NN*CC
  float* kv_part = pw_ws   + (size_t)NN * CC;                  // BB*NH*SEG*1024
  float* km_part = kv_part + (size_t)BB * NH * SEG * HD * HD;  // BB*NH*SEG*32
  float* kv_ws   = km_part + (size_t)BB * NH * SEG * HD;       // BB*NH*1024
  float* km_ws   = kv_ws   + (size_t)BB * NH * HD * HD;        // BB*NH*32

  pw_precompute_kernel<<<NN, 256, 0, stream>>>(posw, pw_ws);
  kv_partial_kernel<<<BB * NH * SEG, 256, 0, stream>>>(x2, x3, pw_ws,
                                                       kv_part, km_part);
  kv_combine_kernel<<<(BB * NH * HD * HD) / 256, 256, 0, stream>>>(kv_part, km_part,
                                                                   kv_ws, km_ws);
  attn_lepe_kernel<<<BB * 256, 256, 0, stream>>>(x1, x3, pw_ws, lw, lb,
                                                 kv_ws, km_ws, out);
}
